// FCLSTMCell_45457933860875
// MI455X (gfx1250) — matrix operand order repaired and checked by reading, not compiled, not run on hardware
//
#include <hip/hip_runtime.h>

typedef __attribute__((ext_vector_type(16))) __bf16 v16bf;
typedef __attribute__((ext_vector_type(8)))  __bf16 v8bf;
typedef __attribute__((ext_vector_type(8)))  float  v8f;
typedef __attribute__((ext_vector_type(4)))  float  v4f;

#define A_CH 64
#define S_CH 256
#define D_CH 256
#define IN_CH 576
#define MID 32
#define OUT_CH 1024

// LDS layout (bytes)
#define W2T_OFF   0                     // bf16 [N=1024][K=32]   = 65536 B
#define W1T_OFF   65536                 // bf16 [N=32][K=576]    = 36864 B
#define H1_OFF    (65536 + 36864)       // bf16 [8 waves][16][32] = 8192 B
#define WRL_OFF   (H1_OFF + 8192)       // f32  [832]            = 3328 B
#define B2L_OFF   (WRL_OFF + 3328)      // f32  [1024]           = 4096 B
#define B1L_OFF   (B2L_OFF + 4096)      // f32  [32]             = 128 B
#define BRL_OFF   (B1L_OFF + 128)       // f32  [4]              = 16 B
#define LDS_BYTES (BRL_OFF + 16)

// sigmoid(x) = rcp(1 + exp(-x));  tanh(x) = 2*sigmoid(2x) - 1  (saturates to +-1)
__device__ __forceinline__ float sigmoidf_(float x) {
    return __builtin_amdgcn_rcpf(1.0f + __expf(-x));
}
__device__ __forceinline__ float tanhf_(float x) {
    float t = __builtin_amdgcn_rcpf(1.0f + __expf(-2.0f * x));
    return __builtin_fmaf(2.0f, t, -1.0f);
}

__device__ __forceinline__ float dot4_(v4f a, v4f b) {
    return a.x*b.x + a.y*b.y + a.z*b.z + a.w*b.w;
}

// One x-segment of GEMM1 (branch-free: segment base pointers precomputed).
// Each K-block kb covers 32 x-columns; this lane supplies A rows via half-runs
// {0..15} (half=0) or {16..31} (half=1), and folds the reward
// x-partial against LDS-resident Wr.
template<int NKB>
__device__ __forceinline__ void gemm1_seg(
    const float* __restrict__ src,   // lane's row base + half*16 within this segment
    const float* __restrict__ wr,    // wrl + segment col base + half*16
    const __bf16* __restrict__ wb0,  // w1t + lm*IN_CH       + cbase + half*16
    const __bf16* __restrict__ wb1,  // w1t + (16+lm)*IN_CH  + cbase + half*16
    v8f& acc0, v8f& acc1, float& rewx)
{
    #pragma unroll 2
    for (int kb = 0; kb < NKB; ++kb) {
        const float* s = src + kb * 32;
        v4f x0 = *(const v4f*)(s);
        v4f x1 = *(const v4f*)(s + 4);
        v4f x2 = *(const v4f*)(s + 8);
        v4f x3 = *(const v4f*)(s + 12);

        const float* w = wr + kb * 32;
        v4f w0 = *(const v4f*)(w);
        v4f w1v = *(const v4f*)(w + 4);
        v4f w2v = *(const v4f*)(w + 8);
        v4f w3v = *(const v4f*)(w + 12);
        rewx += dot4_(x0, w0) + dot4_(x1, w1v) + dot4_(x2, w2v) + dot4_(x3, w3v);

        v16bf a;
        a[0]=(__bf16)x0.x;  a[1]=(__bf16)x0.y;  a[2]=(__bf16)x0.z;  a[3]=(__bf16)x0.w;
        a[4]=(__bf16)x1.x;  a[5]=(__bf16)x1.y;  a[6]=(__bf16)x1.z;  a[7]=(__bf16)x1.w;
        a[8]=(__bf16)x2.x;  a[9]=(__bf16)x2.y;  a[10]=(__bf16)x2.z; a[11]=(__bf16)x2.w;
        a[12]=(__bf16)x3.x; a[13]=(__bf16)x3.y; a[14]=(__bf16)x3.z; a[15]=(__bf16)x3.w;

        v16bf b0  = *(const v16bf*)(wb0 + kb * 32);
        v16bf b1f = *(const v16bf*)(wb1 + kb * 32);

        acc0 = __builtin_amdgcn_wmma_f32_16x16x32_bf16(false, a, false, b0,  (short)0, acc0, false, false);
        acc1 = __builtin_amdgcn_wmma_f32_16x16x32_bf16(false, a, false, b1f, (short)0, acc1, false, false);
    }
}

__global__ __launch_bounds__(256)
__attribute__((amdgpu_waves_per_eu(4)))          // cap at 256 VGPRs: no MSB paging, 4+ waves/SIMD
void fclstm_fused_kernel(
    const float* __restrict__ prev_h, const float* __restrict__ prev_c,
    const float* __restrict__ action, const float* __restrict__ dynamics,
    const float* __restrict__ W1, const float* __restrict__ b1,
    const float* __restrict__ W2, const float* __restrict__ b2,
    const float* __restrict__ Wr, const float* __restrict__ br,
    float* __restrict__ out, int Bn)
{
    extern __shared__ char lds[];
    __bf16* w2t = (__bf16*)(lds + W2T_OFF);   // W2^T  [n][k] = W2[k][n]
    __bf16* w1t = (__bf16*)(lds + W1T_OFF);   // W1^T  [n][k] = W1[k][n]
    __bf16* h1b = (__bf16*)(lds + H1_OFF);    // per-wave h1 tile [16][32]
    float*  wrl = (float*)(lds + WRL_OFF);    // Wr [832]
    float*  b2l = (float*)(lds + B2L_OFF);    // b2 [1024]
    float*  b1l = (float*)(lds + B1L_OFF);    // b1 [32]
    float*  brl = (float*)(lds + BRL_OFF);    // br [1]

    const int tid = threadIdx.x;

    // ---- cooperative constant staging (coalesced global reads, bf16 convert) ----
    for (int i = tid; i < MID * OUT_CH; i += 256) {
        int k = i >> 10, n = i & (OUT_CH - 1);
        w2t[n * MID + k] = (__bf16)W2[i];
    }
    for (int i = tid; i < IN_CH * MID; i += 256) {
        int k = i >> 5, n = i & (MID - 1);
        w1t[n * IN_CH + k] = (__bf16)W1[i];
    }
    for (int i = tid; i < IN_CH + S_CH; i += 256) wrl[i] = Wr[i];
    for (int i = tid; i < OUT_CH; i += 256)       b2l[i] = b2[i];
    if (tid < MID) b1l[tid] = b1[tid];
    if (tid == 0)  brl[0] = br[0];
    __syncthreads();

    const int wave = tid >> 5;
    const int lane = tid & 31;
    const int half = lane >> 4;     // wave32: two 16-lane halves
    const int lm   = lane & 15;
    const int R    = (blockIdx.x * 8 + wave) * 16;   // 16-row tile base
    if (R >= Bn) return;
    const int row  = R + lm;        // A-matrix: lane (mod 16) = row M

    // Prefetch this tile's prev_c rows (consumed in the gate loop) so the
    // HBM fetch overlaps GEMM1:  16 rows x 1KB = 128 cachelines, 4 per lane.
    {
        const float* pcp = prev_c + (size_t)row * S_CH;
        #pragma unroll
        for (int t = 0; t < 4; ++t)
            __builtin_prefetch(pcp + (half * 4 + t) * 32, 0, 0);
    }

    // =================== GEMM1: h1 = tanh(x @ W1 + b1) ===================
    v8f acc0 = {};   // N = 0..15
    v8f acc1 = {};   // N = 16..31
    float rewx = 0.f;

    {
        const int h8 = half * 16;
        const __bf16* wb0 = w1t + (size_t)lm * IN_CH        + half * 16;
        const __bf16* wb1 = w1t + (size_t)(16 + lm) * IN_CH + half * 16;
        // x = [dynamics(256) | action(64) | prev_h(256)]: boundaries are K-block aligned
        gemm1_seg<8>(dynamics + (size_t)row * D_CH + h8, wrl + h8,
                     wb0,       wb1,       acc0, acc1, rewx);
        gemm1_seg<2>(action   + (size_t)row * A_CH + h8, wrl + 256 + h8,
                     wb0 + 256, wb1 + 256, acc0, acc1, rewx);
        gemm1_seg<8>(prev_h   + (size_t)row * S_CH + h8, wrl + 320 + h8,
                     wb0 + 320, wb1 + 320, acc0, acc1, rewx);
    }

    // tanh + re-layout D(16x16) -> A(16x32) through per-wave LDS scratch.
    // D layout: lane lm = col N, vgpr v -> row M = half*8 + v.
    {
        const float bb1a = b1l[lm], bb1b = b1l[16 + lm];
        __bf16* hp = h1b + wave * 512;
        #pragma unroll
        for (int v = 0; v < 8; ++v) {
            int r = half * 8 + v;
            hp[r * 32 + lm]      = (__bf16)tanhf_(acc0[v] + bb1a);
            hp[r * 32 + 16 + lm] = (__bf16)tanhf_(acc1[v] + bb1b);
        }
    }
    // intra-wave ordering only: wait on the CDNA5 DS counter, no barrier needed
    asm volatile("s_wait_dscnt 0x0" ::: "memory");

    v16bf ah;
    {
        const __bf16* hr = h1b + wave * 512 + lm * 32;  // A: lane = row
        v8bf hlo = *(const v8bf*)(hr + half * 16);      // K = half*16 .. +7
        v8bf hhi = *(const v8bf*)(hr + half * 16 + 8);  // K = half*16+8 .. +15
        #pragma unroll
        for (int e = 0; e < 8; ++e) { ah[e] = hlo[e]; ah[8 + e] = hhi[e]; }
    }

    // =================== GEMM2 + LSTM gates + reward ===================
    float rh[8];
    #pragma unroll
    for (int v = 0; v < 8; ++v) rh[v] = 0.f;

    float* out_h = out;
    float* out_c = out + (size_t)Bn * S_CH;
    float* out_r = out + (size_t)2 * Bn * S_CH;

    for (int jb = 0; jb < 16; ++jb) {
        const int n = jb * 16 + lm;              // gate-column for this lane
        const __bf16* q = w2t + half * 16;       // half selects K 0..15 / 16..31
        v16bf bi  = *(const v16bf*)(q + (size_t)(0 * S_CH + n) * MID);
        v16bf bff = *(const v16bf*)(q + (size_t)(1 * S_CH + n) * MID);
        v16bf bc  = *(const v16bf*)(q + (size_t)(2 * S_CH + n) * MID);
        v16bf bo  = *(const v16bf*)(q + (size_t)(3 * S_CH + n) * MID);

        v8f z = {};
        v8f gi = __builtin_amdgcn_wmma_f32_16x16x32_bf16(false, ah, false, bi,  (short)0, z, false, false);
        v8f gf = __builtin_amdgcn_wmma_f32_16x16x32_bf16(false, ah, false, bff, (short)0, z, false, false);
        v8f gc = __builtin_amdgcn_wmma_f32_16x16x32_bf16(false, ah, false, bc,  (short)0, z, false, false);
        v8f go = __builtin_amdgcn_wmma_f32_16x16x32_bf16(false, ah, false, bo,  (short)0, z, false, false);

        const float b2i = b2l[n];
        const float b2f = b2l[S_CH + n];
        const float b2c = b2l[2 * S_CH + n];
        const float b2o = b2l[3 * S_CH + n];
        const float wrn = wrl[IN_CH + n];

        const float* pc  = prev_c + (size_t)(R + half * 8) * S_CH + n;
        float*       oh  = out_h  + (size_t)(R + half * 8) * S_CH + n;
        float*       oc  = out_c  + (size_t)(R + half * 8) * S_CH + n;

        #pragma unroll
        for (int v = 0; v < 8; ++v) {
            float iv = sigmoidf_(gi[v] + b2i);
            float fv = sigmoidf_(gf[v] + b2f);
            float cv = tanhf_(gc[v] + b2c);
            float ov = sigmoidf_(go[v] + b2o);
            float pcv = pc[(size_t)v * S_CH];
            float nc = fv * pcv + iv * cv;
            float nh = ov * tanhf_(nc);
            oc[(size_t)v * S_CH] = nc;      // lanes 0-15 / 16-31 each write 64B contiguous
            oh[(size_t)v * S_CH] = nh;
            rh[v] += nh * wrn;
        }
    }

    // ---- reward reduction: within each 16-lane half (row-wise), + x-part via xor16 ----
    rewx += __shfl_xor(rewx, 16, 32);     // full row sum for row lm, in both halves
    #pragma unroll
    for (int off = 8; off >= 1; off >>= 1) {
        #pragma unroll
        for (int v = 0; v < 8; ++v) rh[v] += __shfl_xor(rh[v], off, 32);
    }

    const float br0 = brl[0];
    float rew[8];
    #pragma unroll
    for (int v = 0; v < 8; ++v) {
        float rx = __shfl(rewx, half * 8 + v, 32);   // lane r (0..15) holds row r's x-part
        rew[v] = tanhf_(rh[v] + rx + br0);
    }
    if (lm == 0) {
        #pragma unroll
        for (int v = 0; v < 8; ++v) out_r[R + half * 8 + v] = rew[v];
    }
}

extern "C" void kernel_launch(void* const* d_in, const int* in_sizes, int n_in,
                              void* d_out, int out_size, void* d_ws, size_t ws_size,
                              hipStream_t stream) {
    const float* prev_h   = (const float*)d_in[0];
    const float* prev_c   = (const float*)d_in[1];
    const float* action   = (const float*)d_in[2];
    const float* dynamics = (const float*)d_in[3];
    const float* W1       = (const float*)d_in[4];
    const float* b1       = (const float*)d_in[5];
    const float* W2       = (const float*)d_in[6];
    const float* b2       = (const float*)d_in[7];
    const float* Wr       = (const float*)d_in[8];
    const float* br       = (const float*)d_in[9];
    float* out = (float*)d_out;

    const int Bn = in_sizes[0] / S_CH;          // 65536
    const int blocks = (Bn + 127) / 128;        // 8 waves/block * 16 rows/wave

    fclstm_fused_kernel<<<blocks, 256, LDS_BYTES, stream>>>(
        prev_h, prev_c, action, dynamics, W1, b1, W2, b2, Wr, br, out, Bn);

    (void)n_in; (void)out_size; (void)d_ws; (void)ws_size;
}
